// MEL_31061203484789
// MI455X (gfx1250) — compile-verified, matrix operand order checked by reading, not run on hardware
//
#include <hip/hip_runtime.h>
#include <hip/hip_bf16.h>

// ---------------------------------------------------------------------------
// Problem constants (from reference)
// ---------------------------------------------------------------------------
constexpr int NWAY   = 5;
constexpr int KSHOT  = 5;
constexpr int Q      = NWAY * KSHOT;   // 25 queries
constexpr int C      = 640;            // channels
constexpr int HW     = 196;            // 14*14 spatial positions
constexpr int MS     = NWAY * HW;      // 980 support columns
constexpr float GAMMA  = 20.0f;
constexpr float GAMMA2 = 10.0f;
constexpr float ALPHA  = 0.5f;
constexpr float EPSN   = 1e-16f;

// Padded dims for WMMA tiling (16x16 tiles, 2x2 tiles per wave => 32 granule)
constexpr int MPAD = 224;   // >= 196 rounded so 32-row wave blocks fit (7 pairs)
constexpr int NPAD = 992;   // 980 -> 62 tiles of 16 -> 31 pairs of 32
constexpr int MPAIRS = 7;   // ceil(13 tiles / 2)
constexpr int NPAIRS = 31;  // 62 / 2
constexpr int KATZ_ITERS = 64;

typedef _Float16 half8  __attribute__((ext_vector_type(8)));
typedef _Float16 v16h   __attribute__((ext_vector_type(16)));
typedef float    v8f    __attribute__((ext_vector_type(8)));

// ---------------------------------------------------------------------------
// Kernel 1: support prototypes  sup[s][c][n] = mean_k feat[25 + s*5 + k][c][n]
// ---------------------------------------------------------------------------
__global__ void k_sup_mean(const float* __restrict__ feat, float* __restrict__ sup) {
    int idx = blockIdx.x * blockDim.x + threadIdx.x;
    const int total = NWAY * C * HW;
    if (idx >= total) return;
    int s = idx / (C * HW);
    int r = idx - s * (C * HW);
    // feat image stride = C*HW
    float acc = 0.0f;
    const float* base = feat + (size_t)(Q + s * KSHOT) * (C * HW) + r;
    #pragma unroll
    for (int k = 0; k < KSHOT; ++k) acc += base[(size_t)k * (C * HW)];
    sup[idx] = acc * (1.0f / KSHOT);
}

// ---------------------------------------------------------------------------
// Kernel 2: inverse norms over channel axis.
//   blocks [0, Q*HW)      : inv_q[q*HW+m]  from feat[25+q][c][m]
//   blocks [Q*HW, +NWAY*HW): inv_s[s*HW+n] from sup[s][c][n]
// ---------------------------------------------------------------------------
__global__ void k_norms(const float* __restrict__ feat, const float* __restrict__ sup,
                        float* __restrict__ inv_q, float* __restrict__ inv_s) {
    __shared__ float red[256];
    int b = blockIdx.x;
    int t = threadIdx.x;
    float acc = 0.0f;
    if (b < Q * HW) {
        int q = b / HW, m = b - q * HW;
        const float* p = feat + (size_t)(Q + q) * (C * HW) + m;
        for (int c = t; c < C; c += 256) { float v = p[(size_t)c * HW]; acc += v * v; }
    } else {
        int bb = b - Q * HW;
        int s = bb / HW, n = bb - s * HW;
        const float* p = sup + (size_t)s * (C * HW) + n;
        for (int c = t; c < C; c += 256) { float v = p[(size_t)c * HW]; acc += v * v; }
    }
    red[t] = acc; __syncthreads();
    for (int off = 128; off > 0; off >>= 1) {
        if (t < off) red[t] += red[t + off];
        __syncthreads();
    }
    if (t == 0) {
        float inv = 1.0f / (EPSN + sqrtf(red[0]));
        if (b < Q * HW) inv_q[b] = inv; else inv_s[b - Q * HW] = inv;
    }
}

// ---------------------------------------------------------------------------
// Kernel 3a: pack A (f16): A16[q][m][c] = feat[25+q][c][m] * inv_q, m padded
// ---------------------------------------------------------------------------
__global__ void k_packA(const float* __restrict__ feat, const float* __restrict__ inv_q,
                        _Float16* __restrict__ A16) {
    int idx = blockIdx.x * blockDim.x + threadIdx.x;
    const int total = Q * MPAD * C;
    if (idx >= total) return;
    int q = idx / (MPAD * C);
    int r = idx - q * (MPAD * C);
    int m = r / C;
    int c = r - m * C;
    float v = 0.0f;
    if (m < HW)
        v = feat[((size_t)(Q + q) * C + c) * HW + m] * inv_q[q * HW + m];
    A16[idx] = (_Float16)v;
}

// ---------------------------------------------------------------------------
// Kernel 3b: pack B transposed (f16): BT16[N][c] = sup[s][c][n]*inv_s, N=s*HW+n
// ---------------------------------------------------------------------------
__global__ void k_packB(const float* __restrict__ sup, const float* __restrict__ inv_s,
                        _Float16* __restrict__ BT16) {
    int idx = blockIdx.x * blockDim.x + threadIdx.x;
    const int total = NPAD * C;
    if (idx >= total) return;
    int N = idx / C;
    int c = idx - N * C;
    float v = 0.0f;
    if (N < MS) {
        int s = N / HW, n = N - s * HW;
        v = sup[((size_t)s * C + c) * HW + n] * inv_s[N];
    }
    BT16[idx] = (_Float16)v;
}

// ---------------------------------------------------------------------------
// Kernel 4: WMMA GEMM. S[q] (196x980) = A[q] (196x640) * B (640x980)
// One wave computes a 32x32 output block (2x2 tiles of 16x16), K-loop step 32.
// Fragment loads: plain global_load_b128 per lane (row-major memory, ISA 10.9):
//   A lane addr: row = m0 + (lane&15), halfcol = k0 + 8*(lane>>4)  [+16 for hi8]
//   B stored transposed [N][K] so it uses the identical pattern with row = n.
// ---------------------------------------------------------------------------
__device__ __forceinline__ v16h frag_ld(const _Float16* p) {
    half8 lo = *(const half8*)(p);
    half8 hi = *(const half8*)(p + 16);
    return __builtin_shufflevector(lo, hi, 0,1,2,3,4,5,6,7,8,9,10,11,12,13,14,15);
}

__global__ __launch_bounds__(128) void k_gemm_wmma(const _Float16* __restrict__ A16,
                                                   const _Float16* __restrict__ BT16,
                                                   float* __restrict__ S) {
    const int lane = threadIdx.x & 31;
    const int wave = threadIdx.x >> 5;
    int task = blockIdx.x * 4 + wave;                    // one 32x32 block per wave
    const int ntasks = Q * MPAIRS * NPAIRS;              // 25*7*31 = 5425
    if (task >= ntasks) return;

    int q  = task / (MPAIRS * NPAIRS);
    int r  = task - q * (MPAIRS * NPAIRS);
    int mp = r / NPAIRS;
    int np = r - mp * NPAIRS;
    const int m0 = mp * 32;
    const int n0 = np * 32;

    const int rsel = lane & 15;
    const int hcol = 8 * (lane >> 4);

    const _Float16* a0p = A16 + ((size_t)q * MPAD + (m0 + rsel)) * C + hcol;
    const _Float16* a1p = a0p + (size_t)16 * C;
    const _Float16* b0p = BT16 + (size_t)(n0 + rsel) * C + hcol;
    const _Float16* b1p = b0p + (size_t)16 * C;

    v8f c00 = {}, c01 = {}, c10 = {}, c11 = {};
    #pragma unroll 4
    for (int k = 0; k < C; k += 32) {
        v16h a0 = frag_ld(a0p + k);
        v16h a1 = frag_ld(a1p + k);
        v16h b0 = frag_ld(b0p + k);
        v16h b1 = frag_ld(b1p + k);
        c00 = __builtin_amdgcn_wmma_f32_16x16x32_f16(false, a0, false, b0, (short)0, c00, false, false);
        c01 = __builtin_amdgcn_wmma_f32_16x16x32_f16(false, a0, false, b1, (short)0, c01, false, false);
        c10 = __builtin_amdgcn_wmma_f32_16x16x32_f16(false, a1, false, b0, (short)0, c10, false, false);
        c11 = __builtin_amdgcn_wmma_f32_16x16x32_f16(false, a1, false, b1, (short)0, c11, false, false);
    }

    // C/D layout: VGPR v <-> (M = v + 8*(lane>>4), N = lane&15) within 16x16 tile
    const int mrow = 8 * (lane >> 4);
    const int ncol = lane & 15;
    float* Sq = S + (size_t)q * HW * MS;
    #pragma unroll
    for (int v = 0; v < 8; ++v) {
        int mA = m0 + mrow + v;
        int mB = mA + 16;
        int nA = n0 + ncol;
        int nB = nA + 16;
        if (mA < HW) {
            if (nA < MS) Sq[(size_t)mA * MS + nA] = c00[v];
            if (nB < MS) Sq[(size_t)mA * MS + nB] = c01[v];
        }
        if (mB < HW) {
            if (nA < MS) Sq[(size_t)mB * MS + nA] = c10[v];
            if (nB < MS) Sq[(size_t)mB * MS + nB] = c11[v];
        }
    }
}

// ---------------------------------------------------------------------------
// Kernel 5a: T_sq = softmax(GAMMA * S) over last axis (980). Block per (q,m).
// ---------------------------------------------------------------------------
__global__ void k_softmax_row(const float* __restrict__ S, float* __restrict__ Tsq) {
    __shared__ float red[256];
    int b = blockIdx.x;            // q*HW + m
    int t = threadIdx.x;
    const float* row = S + (size_t)b * MS;
    float mx = -3.0e38f;
    for (int j = t; j < MS; j += 256) mx = fmaxf(mx, row[j]);
    red[t] = mx; __syncthreads();
    for (int off = 128; off > 0; off >>= 1) { if (t < off) red[t] = fmaxf(red[t], red[t + off]); __syncthreads(); }
    float M = red[0]; __syncthreads();
    float acc = 0.0f;
    for (int j = t; j < MS; j += 256) acc += __expf(GAMMA * (row[j] - M));
    red[t] = acc; __syncthreads();
    for (int off = 128; off > 0; off >>= 1) { if (t < off) red[t] += red[t + off]; __syncthreads(); }
    float inv = 1.0f / red[0];
    float* out = Tsq + (size_t)b * MS;
    for (int j = t; j < MS; j += 256) out[j] = __expf(GAMMA * (row[j] - M)) * inv;
}

// ---------------------------------------------------------------------------
// Kernel 5b: T_qs = softmax(GAMMA2 * S^T) over 196 axis. Block per (q,j).
// Output layout Tqs[q][j][m] (contiguous in m).
// ---------------------------------------------------------------------------
__global__ void k_softmax_col(const float* __restrict__ S, float* __restrict__ Tqs) {
    __shared__ float red[128];
    int b = blockIdx.x;            // q*MS + j
    int q = b / MS, j = b - q * MS;
    int t = threadIdx.x;
    const float* col = S + (size_t)q * HW * MS + j;   // stride MS over m
    float mx = -3.0e38f;
    for (int m = t; m < HW; m += 128) mx = fmaxf(mx, col[(size_t)m * MS]);
    red[t] = mx; __syncthreads();
    for (int off = 64; off > 0; off >>= 1) { if (t < off) red[t] = fmaxf(red[t], red[t + off]); __syncthreads(); }
    float M = red[0]; __syncthreads();
    float acc = 0.0f;
    for (int m = t; m < HW; m += 128) acc += __expf(GAMMA2 * (col[(size_t)m * MS] - M));
    red[t] = acc; __syncthreads();
    for (int off = 64; off > 0; off >>= 1) { if (t < off) red[t] += red[t + off]; __syncthreads(); }
    float inv = 1.0f / red[0];
    float* out = Tqs + (size_t)b * HW;
    for (int m = t; m < HW; m += 128) out[m] = __expf(GAMMA2 * (col[(size_t)m * MS] - M)) * inv;
}

// ---------------------------------------------------------------------------
// Kernel 6: Katz solve per query via fixed-point on the block structure
//   x_s = 1 + a * A x_q   (A[i,j] = Tsq[q][j][i])
//   x_q = 1 + a * B x_s   (B[j,i] = Tqs[q][i][j])
// then partial = x_s - 1, normalize, sum per way -> d_out[q][way].
// Coalesced: consecutive threads read consecutive columns of Tsq/Tqs.
// ---------------------------------------------------------------------------
__global__ __launch_bounds__(1024) void k_katz(const float* __restrict__ Tsq,
                                               const float* __restrict__ Tqs,
                                               float* __restrict__ out) {
    __shared__ float xs[MS];
    __shared__ float xq[HW];
    __shared__ float sums[NWAY];
    int q = blockIdx.x;
    int t = threadIdx.x;

    if (t < MS) xs[t] = 1.0f;
    if (t < HW) xq[t] = 1.0f;
    __syncthreads();

    const float* Aq = Tsq + (size_t)q * HW * MS;   // [j][i]
    const float* Bq = Tqs + (size_t)q * MS * HW;   // [i][j]

    for (int it = 0; it < KATZ_ITERS; ++it) {
        float xs_new = 0.0f;
        if (t < MS) {
            float acc = 0.0f;
            for (int j = 0; j < HW; ++j) acc = fmaf(Aq[(size_t)j * MS + t], xq[j], acc);
            xs_new = 1.0f + ALPHA * acc;
        }
        __syncthreads();
        if (t < MS) xs[t] = xs_new;
        __syncthreads();
        float xq_new = 0.0f;
        if (t < HW) {
            float acc = 0.0f;
            for (int i = 0; i < MS; ++i) acc = fmaf(Bq[(size_t)i * HW + t], xs[i], acc);
            xq_new = 1.0f + ALPHA * acc;
        }
        __syncthreads();
        if (t < HW) xq[t] = xq_new;
        __syncthreads();
    }

    if (t < NWAY) sums[t] = 0.0f;
    __syncthreads();
    if (t < MS) atomicAdd(&sums[t / HW], xs[t] - 1.0f);
    __syncthreads();
    if (t < NWAY) {
        float tot = sums[0] + sums[1] + sums[2] + sums[3] + sums[4];
        out[q * NWAY + t] = sums[t] / tot;
    }
}

// ---------------------------------------------------------------------------
// Host-side launcher
// ---------------------------------------------------------------------------
static inline size_t align256(size_t x) { return (x + 255) & ~size_t(255); }

extern "C" void kernel_launch(void* const* d_in, const int* in_sizes, int n_in,
                              void* d_out, int out_size, void* d_ws, size_t ws_size,
                              hipStream_t stream) {
    (void)in_sizes; (void)n_in; (void)out_size; (void)ws_size;
    const float* feat = (const float*)d_in[0];
    // d_in[1] (query_y) is unused by the reference output.
    float* out = (float*)d_out;

    char* ws = (char*)d_ws;
    size_t o = 0;
    float*    sup   = (float*)(ws + o);  o = align256(o + (size_t)NWAY * C * HW * 4);
    float*    inv_q = (float*)(ws + o);  o = align256(o + (size_t)Q * HW * 4);
    float*    inv_s = (float*)(ws + o);  o = align256(o + (size_t)NWAY * HW * 4);
    _Float16* A16   = (_Float16*)(ws + o); o = align256(o + (size_t)Q * MPAD * C * 2);
    _Float16* BT16  = (_Float16*)(ws + o); o = align256(o + (size_t)NPAD * C * 2);
    float*    Smat  = (float*)(ws + o);  o = align256(o + (size_t)Q * HW * MS * 4);
    float*    Tsq   = (float*)(ws + o);  o = align256(o + (size_t)Q * HW * MS * 4);
    float*    Tqs   = (float*)(ws + o);  o = align256(o + (size_t)Q * MS * HW * 4);

    // 1: support prototypes (mean over shots)
    {
        int total = NWAY * C * HW;
        k_sup_mean<<<(total + 255) / 256, 256, 0, stream>>>(feat, sup);
    }
    // 2: inverse channel norms for query positions and support positions
    k_norms<<<Q * HW + NWAY * HW, 256, 0, stream>>>(feat, sup, inv_q, inv_s);
    // 3: pack f16 operands (A row-major [m][c], B transposed row-major [N][c])
    {
        int totalA = Q * MPAD * C;
        k_packA<<<(totalA + 255) / 256, 256, 0, stream>>>(feat, inv_q, A16);
        int totalB = NPAD * C;
        k_packB<<<(totalB + 255) / 256, 256, 0, stream>>>(sup, inv_s, BT16);
    }
    // 4: WMMA batched GEMM -> S (25 x 196 x 980), 6.1 GFLOP in f16 WMMA
    {
        int ntasks = Q * MPAIRS * NPAIRS;        // 5425 waves, 4 waves/block
        k_gemm_wmma<<<(ntasks + 3) / 4, 128, 0, stream>>>(A16, BT16, Smat);
    }
    // 5: softmaxes
    k_softmax_row<<<Q * HW, 256, 0, stream>>>(Smat, Tsq);
    k_softmax_col<<<Q * MS, 128, 0, stream>>>(Smat, Tqs);
    // 6: Katz fixed-point solve + per-way aggregation -> d_out (25 x 5)
    k_katz<<<Q, 1024, 0, stream>>>(Tsq, Tqs, out);
}
